// Pointnet2Backbone_5669356832930
// MI455X (gfx1250) — matrix operand order repaired and checked by reading, not compiled, not run on hardware
//
#include <hip/hip_runtime.h>
#include <hip/hip_bf16.h>

typedef __attribute__((ext_vector_type(8)))  _Float16 v8h;
typedef __attribute__((ext_vector_type(16))) _Float16 v16h;
typedef __attribute__((ext_vector_type(8)))  float    v8f;

#define BATCH 4
#define BN_EPS 1e-5f

// ------------------------------------------------------------------
// split pointcloud (B,N,4) -> xyz (B,N,3) and features (B,1,N)
// ------------------------------------------------------------------
__global__ void k_split(const float* __restrict__ pc, float* __restrict__ xyz,
                        float* __restrict__ f0, int N) {
  int t = blockIdx.x * blockDim.x + threadIdx.x;
  if (t >= BATCH * N) return;
  const float* p = pc + (size_t)t * 4;
  xyz[(size_t)t * 3 + 0] = p[0];
  xyz[(size_t)t * 3 + 1] = p[1];
  xyz[(size_t)t * 3 + 2] = p[2];
  f0[t] = p[3];
}

// ------------------------------------------------------------------
// Furthest point sampling: one block per batch, serial argmax steps.
// ------------------------------------------------------------------
__global__ __launch_bounds__(256) void k_fps(const float* __restrict__ xyz,
                                             int N, int npoint,
                                             int* __restrict__ inds,
                                             float* __restrict__ dist) {
  int b = blockIdx.x;
  const float* P = xyz + (size_t)b * N * 3;
  float* D = dist + (size_t)b * N;
  __shared__ float s_val[256];
  __shared__ int   s_idx[256];
  __shared__ float s_p[3];
  for (int i = threadIdx.x; i < N; i += 256) D[i] = 1e10f;
  if (threadIdx.x == 0) inds[(size_t)b * npoint] = 0;
  int last = 0;
  __syncthreads();
  for (int t = 1; t < npoint; ++t) {
    if (threadIdx.x == 0) {
      s_p[0] = P[(size_t)last * 3 + 0];
      s_p[1] = P[(size_t)last * 3 + 1];
      s_p[2] = P[(size_t)last * 3 + 2];
    }
    __syncthreads();
    float px = s_p[0], py = s_p[1], pz = s_p[2];
    float best = -1.f; int bi = N;
    for (int i = threadIdx.x; i < N; i += 256) {
      float dx = P[(size_t)i * 3 + 0] - px;
      float dy = P[(size_t)i * 3 + 1] - py;
      float dz = P[(size_t)i * 3 + 2] - pz;
      float d = dx * dx + dy * dy + dz * dz;
      float m = fminf(D[i], d);
      D[i] = m;
      if (m > best) { best = m; bi = i; }
    }
    s_val[threadIdx.x] = best; s_idx[threadIdx.x] = bi;
    __syncthreads();
    for (int s = 128; s > 0; s >>= 1) {
      if (threadIdx.x < (unsigned)s) {
        float ov = s_val[threadIdx.x + s]; int oi = s_idx[threadIdx.x + s];
        if (ov > s_val[threadIdx.x] ||
            (ov == s_val[threadIdx.x] && oi < s_idx[threadIdx.x])) {
          s_val[threadIdx.x] = ov; s_idx[threadIdx.x] = oi;
        }
      }
      __syncthreads();
    }
    last = s_idx[0];
    if (threadIdx.x == 0) inds[(size_t)b * npoint + t] = last;
    __syncthreads();
  }
}

__global__ void k_gather_xyz(const float* __restrict__ xyz, const int* __restrict__ inds,
                             int Nin, int np, float* __restrict__ nxyz) {
  int t = blockIdx.x * blockDim.x + threadIdx.x;
  if (t >= BATCH * np) return;
  int b = t / np;
  int i = inds[t];
  const float* p = xyz + ((size_t)b * Nin + i) * 3;
  nxyz[(size_t)t * 3 + 0] = p[0];
  nxyz[(size_t)t * 3 + 1] = p[1];
  nxyz[(size_t)t * 3 + 2] = p[2];
}

// ------------------------------------------------------------------
// Ball query: in-order scan == nsample smallest in-radius indices.
// ------------------------------------------------------------------
__global__ void k_ballquery(const float* __restrict__ xyz, const float* __restrict__ nxyz,
                            int N, int np, int ns, float r2, int* __restrict__ idx) {
  int t = blockIdx.x * blockDim.x + threadIdx.x;
  if (t >= BATCH * np) return;
  int b = t / np;
  const float* P = xyz + (size_t)b * N * 3;
  float qx = nxyz[(size_t)t * 3 + 0];
  float qy = nxyz[(size_t)t * 3 + 1];
  float qz = nxyz[(size_t)t * 3 + 2];
  int* out = idx + (size_t)t * ns;
  int cnt = 0, first = 0; bool has = false;
  for (int i = 0; i < N && cnt < ns; ++i) {
    float dx = P[(size_t)i * 3 + 0] - qx;
    float dy = P[(size_t)i * 3 + 1] - qy;
    float dz = P[(size_t)i * 3 + 2] - qz;
    float d = dx * dx + dy * dy + dz * dz;
    if (d < r2) {
      if (!has) { first = i; has = true; }
      out[cnt++] = i;
    }
  }
  for (; cnt < ns; ++cnt) out[cnt] = first;
}

// ------------------------------------------------------------------
// Build f16 GEMM input for an SA module, K-major per column.
// ------------------------------------------------------------------
__global__ void k_build_x_sa(const float* __restrict__ xyz, const float* __restrict__ nxyz,
                             const float* __restrict__ F, const int* __restrict__ idx,
                             int N, int np, int ns, int Cin, int Kp, float inv_r,
                             _Float16* __restrict__ X) {
  long col = (long)blockIdx.x * blockDim.x + threadIdx.x;
  long tot = (long)BATCH * np * ns;
  if (col >= tot) return;
  long rem = col / ns;
  int j = (int)(rem % np);
  int b = (int)(rem / np);
  int i = idx[col];
  const float* q = nxyz + ((size_t)b * np + j) * 3;
  const float* p = xyz + ((size_t)b * N + i) * 3;
  _Float16* xc = X + (size_t)col * Kp;
  xc[0] = (_Float16)((p[0] - q[0]) * inv_r);
  xc[1] = (_Float16)((p[1] - q[1]) * inv_r);
  xc[2] = (_Float16)((p[2] - q[2]) * inv_r);
  for (int c = 0; c < Cin; ++c)
    xc[3 + c] = (_Float16)F[((size_t)b * Cin + c) * N + i];
  for (int k = 3 + Cin; k < Kp; ++k) xc[k] = (_Float16)0.f;
}

// ------------------------------------------------------------------
// FP module: 3-NN interpolation + concat -> f16 GEMM input.
// ------------------------------------------------------------------
__global__ void k_build_x_fp(const float* __restrict__ uxyz, const float* __restrict__ kxyz,
                             const float* __restrict__ UF, const float* __restrict__ KF,
                             int nu, int nk, int Cu, int Ck, _Float16* __restrict__ X) {
  int t = blockIdx.x * blockDim.x + threadIdx.x;
  if (t >= BATCH * nu) return;
  int b = t / nu, j = t % nu;
  const float* q = uxyz + (size_t)t * 3;
  const float* P = kxyz + (size_t)b * nk * 3;
  float d0 = 1e30f, d1 = 1e30f, d2 = 1e30f;
  int i0 = 0, i1 = 0, i2 = 0;
  for (int i = 0; i < nk; ++i) {
    float dx = P[(size_t)i * 3 + 0] - q[0];
    float dy = P[(size_t)i * 3 + 1] - q[1];
    float dz = P[(size_t)i * 3 + 2] - q[2];
    float d = dx * dx + dy * dy + dz * dz;
    if (d < d0)      { d2 = d1; i2 = i1; d1 = d0; i1 = i0; d0 = d; i0 = i; }
    else if (d < d1) { d2 = d1; i2 = i1; d1 = d; i1 = i; }
    else if (d < d2) { d2 = d; i2 = i; }
  }
  float r0 = 1.f / (d0 + 1e-8f);
  float r1 = 1.f / (d1 + 1e-8f);
  float r2 = 1.f / (d2 + 1e-8f);
  float inv = 1.f / (r0 + r1 + r2);
  float w0 = r0 * inv, w1 = r1 * inv, w2 = r2 * inv;
  _Float16* xc = X + (size_t)t * (size_t)(Ck + Cu);
  const float* Kb = KF + (size_t)b * Ck * nk;
  for (int c = 0; c < Ck; ++c) {
    const float* row = Kb + (size_t)c * nk;
    xc[c] = (_Float16)(w0 * row[i0] + w1 * row[i1] + w2 * row[i2]);
  }
  const float* Ub = UF + (size_t)b * Cu * nu;
  for (int c = 0; c < Cu; ++c) xc[Ck + c] = (_Float16)Ub[(size_t)c * nu + j];
}

// ------------------------------------------------------------------
// Convert weights W[O][C] f32 -> Wh[O][Kp] f16 (zero padded K).
// ------------------------------------------------------------------
__global__ void k_cvt_w(const float* __restrict__ W, _Float16* __restrict__ Wh,
                        int O, int C, int Kp) {
  int t = blockIdx.x * blockDim.x + threadIdx.x;
  if (t >= O * Kp) return;
  int o = t / Kp, k = t % Kp;
  Wh[t] = (k < C) ? (_Float16)W[(size_t)o * C + k] : (_Float16)0.f;
}

// ------------------------------------------------------------------
// Register-blocked WMMA GEMM with fused BN partial stats.
//   Y[M][Ncols] = Wh[M][Kp] * X[Ncols][Kp]^T
// Block = 128 threads = 4 waves; grid = (Ncols/64, M/64).
// Each wave owns one 16-column strip (n0) and all 64 rows of the
// M-tile: 4 WMMA accumulators reuse one B fragment per k-step (4 WMMA
// per B load). A fragments are shared across waves/blocks -> L1/L2.
// Epilogue: accumulators staged to LDS; per-row (sum, sum^2) over the
// block's 64 columns written to psum/psum2[M][Ncols/64] (deterministic
// two-stage BN reduction, avoids re-reading the f32 Y from HBM).
// Frag layouts per CDNA5 ISA 7.12.2:
//   A: lane(l,h) row=m, halves k={8h..8h+7, 16+8h..16+8h+7}
//   B: lane(l,h) col=n, halves k={8h..8h+7, 16+8h..16+8h+7}
//   D: VGPR r -> (row r+8h, col l)
// ------------------------------------------------------------------
__global__ __launch_bounds__(128) void k_wmma_gemm(const _Float16* __restrict__ A,
                                                   const _Float16* __restrict__ Xm,
                                                   float* __restrict__ Y,
                                                   float* __restrict__ psum,
                                                   float* __restrict__ psum2,
                                                   int M, int Kp, long Ncols) {
  const int lane = threadIdx.x & 31;
  const int wave = threadIdx.x >> 5;
  const int h = lane >> 4;
  const int l = lane & 15;
  const long n0 = ((long)blockIdx.x * 4 + wave) * 16;
  const int m0 = blockIdx.y * 64;
  const _Float16* Bp = Xm + (size_t)(n0 + l) * Kp;

  v8f acc[4] = {};
  for (int k0 = 0; k0 < Kp; k0 += 32) {
    v8h blo = *(const v8h*)(Bp + k0 + 8 * h);
    v8h bhi = *(const v8h*)(Bp + k0 + 16 + 8 * h);
    v16h bm;
#pragma unroll
    for (int e = 0; e < 8; ++e) { bm[e] = blo[e]; bm[e + 8] = bhi[e]; }
    __builtin_prefetch(Bp + k0 + 64, 0, 1);
#pragma unroll
    for (int mt = 0; mt < 4; ++mt) {
      const _Float16* Ap = A + (size_t)(m0 + mt * 16 + l) * Kp + k0 + 8 * h;
      v8h alo = *(const v8h*)(Ap);
      v8h ahi = *(const v8h*)(Ap + 16);
      v16h a;
#pragma unroll
      for (int e = 0; e < 8; ++e) { a[e] = alo[e]; a[e + 8] = ahi[e]; }
      acc[mt] = __builtin_amdgcn_wmma_f32_16x16x32_f16(false, a, false, bm,
                                                       (short)0, acc[mt], false, false);
    }
  }

  // store D tiles
#pragma unroll
  for (int mt = 0; mt < 4; ++mt) {
    float* yp = Y + (size_t)(m0 + mt * 16) * Ncols + n0 + l;
#pragma unroll
    for (int r = 0; r < 8; ++r)
      yp[(size_t)(r + 8 * h) * Ncols] = acc[mt][r];
  }

  // fused BN partial stats: per-row sums over this block's 64 columns
  __shared__ float tile[4][64][16];   // [wave][local row][local col]
#pragma unroll
  for (int mt = 0; mt < 4; ++mt)
#pragma unroll
    for (int r = 0; r < 8; ++r)
      tile[wave][mt * 16 + 8 * h + r][l] = acc[mt][r];
  __syncthreads();
  if (threadIdx.x < 64) {
    int i = threadIdx.x;           // local row
    float s = 0.f, s2 = 0.f;
#pragma unroll
    for (int w = 0; w < 4; ++w)
      for (int c = 0; c < 16; ++c) {
        float v = tile[w][i][c];
        s += v; s2 += v * v;
      }
    long NB = Ncols >> 6;          // Ncols / 64
    psum [(size_t)(m0 + i) * NB + blockIdx.x] = s;
    psum2[(size_t)(m0 + i) * NB + blockIdx.x] = s2;
  }
}

// ------------------------------------------------------------------
// Stage-2 BN stats: reduce per-block partials (fixed order; no atomics).
// ------------------------------------------------------------------
__global__ __launch_bounds__(256) void k_bnstats2(const float* __restrict__ psum,
                                                  const float* __restrict__ psum2,
                                                  long NB, long cols,
                                                  float* __restrict__ mean,
                                                  float* __restrict__ var) {
  int m = blockIdx.x;
  const float* r1 = psum + (size_t)m * NB;
  const float* r2 = psum2 + (size_t)m * NB;
  __shared__ float s1[256], s2[256];
  float a = 0.f, b = 0.f;
  for (long i = threadIdx.x; i < NB; i += 256) { a += r1[i]; b += r2[i]; }
  s1[threadIdx.x] = a; s2[threadIdx.x] = b;
  __syncthreads();
  for (int s = 128; s > 0; s >>= 1) {
    if (threadIdx.x < (unsigned)s) {
      s1[threadIdx.x] += s1[threadIdx.x + s];
      s2[threadIdx.x] += s2[threadIdx.x + s];
    }
    __syncthreads();
  }
  if (threadIdx.x == 0) {
    float mu = s1[0] / (float)cols;
    mean[m] = mu;
    var[m] = s2[0] / (float)cols - mu * mu;
  }
}

__device__ __forceinline__ float bn_act(float y, float mu, float vv, float g, float be) {
  float v = (y - mu) * rsqrtf(vv + BN_EPS) * g + be;
  return v > 0.f ? v : 0.f;
}

// BN+ReLU -> next layer f16 input (K-major per column, Kp_next == O)
__global__ void k_bn_relu_to_x(const float* __restrict__ Y,
                               const float* __restrict__ mean, const float* __restrict__ var,
                               const float* __restrict__ gamma, const float* __restrict__ beta,
                               _Float16* __restrict__ X, int O, long cols) {
  long t = (long)blockIdx.x * blockDim.x + threadIdx.x;
  if (t >= (long)O * cols) return;
  int m = (int)(t % O);
  long col = t / O;
  float v = bn_act(Y[(size_t)m * cols + col], mean[m], var[m], gamma[m], beta[m]);
  X[(size_t)col * O + m] = (_Float16)v;
}

// BN+ReLU then max over nsample -> SA output features (B, O, np)
__global__ void k_bn_relu_max(const float* __restrict__ Y,
                              const float* __restrict__ mean, const float* __restrict__ var,
                              const float* __restrict__ gamma, const float* __restrict__ beta,
                              float* __restrict__ Fout, int O, int np, int ns) {
  long t = (long)blockIdx.x * blockDim.x + threadIdx.x;
  long tot = (long)BATCH * O * np;
  if (t >= tot) return;
  int j = (int)(t % np);
  long r = t / np;
  int m = (int)(r % O);
  int b = (int)(r / O);
  long cols = (long)BATCH * np * ns;
  const float* row = Y + (size_t)m * cols + ((size_t)b * np + j) * ns;
  float mu = mean[m], vv = var[m], g = gamma[m], be = beta[m];
  float inv = rsqrtf(vv + BN_EPS);
  float best = -1e30f;
  for (int s = 0; s < ns; ++s) {
    float v = (row[s] - mu) * inv * g + be;
    v = v > 0.f ? v : 0.f;
    best = fmaxf(best, v);
  }
  Fout[((size_t)b * O + m) * np + j] = best;
}

// BN+ReLU -> FP output features (B, O, nu) f32
__global__ void k_bn_relu_fp_out(const float* __restrict__ Y,
                                 const float* __restrict__ mean, const float* __restrict__ var,
                                 const float* __restrict__ gamma, const float* __restrict__ beta,
                                 float* __restrict__ Fout, int O, int nu, long cols) {
  long t = (long)blockIdx.x * blockDim.x + threadIdx.x;
  if (t >= (long)O * cols) return;
  int m = (int)(t % O);
  long col = t / O;
  int b = (int)(col / nu);
  int j = (int)(col % nu);
  float v = bn_act(Y[(size_t)m * cols + col], mean[m], var[m], gamma[m], beta[m]);
  Fout[((size_t)b * O + m) * nu + j] = v;
}

// Copy fp2_xyz and fp2_inds into the output tail.
__global__ void k_tail(const float* __restrict__ xyz2, const int* __restrict__ inds1,
                       float* __restrict__ out) {
  const int FEAT = BATCH * 256 * 1024;   // 1048576
  const int NX = BATCH * 1024 * 3;       // 12288
  int t = blockIdx.x * blockDim.x + threadIdx.x;
  if (t < NX) { out[FEAT + t] = xyz2[t]; return; }
  int u = t - NX;
  if (u < BATCH * 1024) {
    int b = u >> 10, j = u & 1023;
    out[FEAT + NX + u] = (float)inds1[(size_t)b * 2048 + j];
  }
}

// ==================================================================
extern "C" void kernel_launch(void* const* d_in, const int* in_sizes, int n_in,
                              void* d_out, int out_size, void* d_ws, size_t ws_size,
                              hipStream_t stream) {
  (void)in_sizes; (void)n_in; (void)out_size; (void)ws_size;
  const int N0 = 16384;
  const float* pc = (const float*)d_in[0];

  // Input layout: dict order {pointcloud, params}; params pytree-sorted:
  // fp1, fp2, sa1, sa2, sa3, sa4; per layer keys sorted: W, beta, gamma.
  const int fp1b = 1, fp2b = 7, sa1b = 13, sa2b = 22, sa3b = 31, sa4b = 40;
  auto Wp = [&](int base, int l) { return (const float*)d_in[base + 3 * l + 0]; };
  auto Bq = [&](int base, int l) { return (const float*)d_in[base + 3 * l + 1]; };
  auto Gp = [&](int base, int l) { return (const float*)d_in[base + 3 * l + 2]; };

  // ---- workspace bump allocator (256B aligned) ----
  char* ws = (char*)d_ws;
  size_t off = 0;
  auto alloc = [&](size_t bytes) -> void* {
    off = (off + 255) & ~(size_t)255;
    void* p = ws + off;
    off += bytes;
    return p;
  };
  float* xyz0 = (float*)alloc((size_t)BATCH * N0 * 3 * 4);
  float* f0   = (float*)alloc((size_t)BATCH * N0 * 4);
  float* xyz1 = (float*)alloc((size_t)BATCH * 2048 * 3 * 4);
  float* f1   = (float*)alloc((size_t)BATCH * 128 * 2048 * 4);
  float* xyz2 = (float*)alloc((size_t)BATCH * 1024 * 3 * 4);
  float* f2   = (float*)alloc((size_t)BATCH * 256 * 1024 * 4);
  float* xyz3 = (float*)alloc((size_t)BATCH * 512 * 3 * 4);
  float* f3   = (float*)alloc((size_t)BATCH * 256 * 512 * 4);
  float* xyz4 = (float*)alloc((size_t)BATCH * 256 * 3 * 4);
  float* f4   = (float*)alloc((size_t)BATCH * 256 * 256 * 4);
  float* fp1o = (float*)alloc((size_t)BATCH * 256 * 512 * 4);
  int* inds1  = (int*)alloc((size_t)BATCH * 2048 * 4);
  int* indsT  = (int*)alloc((size_t)BATCH * 2048 * 4);
  float* distb = (float*)alloc((size_t)BATCH * N0 * 4);
  int* idxb   = (int*)alloc((size_t)BATCH * 2048 * 64 * 4);
  _Float16* Wh = (_Float16*)alloc((size_t)256 * 512 * 2);
  float* meanb = (float*)alloc(256 * 4);
  float* varb  = (float*)alloc(256 * 4);
  float* psumb  = (float*)alloc((size_t)256 * 8192 * 4);        // [M][Ncols/64] max
  float* psum2b = (float*)alloc((size_t)256 * 8192 * 4);
  _Float16* Xbuf = (_Float16*)alloc((size_t)524288 * 64 * 2);   // 67 MB
  float* Ybuf    = (float*)alloc((size_t)128 * 524288 * 4);     // 268 MB

  // ---- per-layer pipeline: f16 weights -> WMMA GEMM(+stats) -> stage2 ----
  auto mlp_layer = [&](const float* W, int O, int C, long cols) {
    int Kp = (C + 31) & ~31;
    int tot = O * Kp;
    k_cvt_w<<<(tot + 255) / 256, 256, 0, stream>>>(W, Wh, O, C, Kp);
    k_wmma_gemm<<<dim3((unsigned)(cols / 64), O / 64), 128, 0, stream>>>(
        Wh, Xbuf, Ybuf, psumb, psum2b, O, Kp, cols);
    k_bnstats2<<<O, 256, 0, stream>>>(psumb, psum2b, cols / 64, cols, meanb, varb);
  };
  auto to_x = [&](const float* gamma, const float* beta, int O, long cols) {
    long tot = (long)O * cols;
    k_bn_relu_to_x<<<(unsigned)((tot + 255) / 256), 256, 0, stream>>>(
        Ybuf, meanb, varb, gamma, beta, Xbuf, O, cols);
  };

  auto run_sa = [&](const float* xin, int Nin, const float* Fin, int Cin,
                    int np, int ns, float rad, int wb, const int* d4,
                    float* xout, float* Fout, int* inds) {
    k_fps<<<BATCH, 256, 0, stream>>>(xin, Nin, np, inds, distb);
    int tq = BATCH * np;
    k_gather_xyz<<<(tq + 255) / 256, 256, 0, stream>>>(xin, inds, Nin, np, xout);
    k_ballquery<<<(tq + 255) / 256, 256, 0, stream>>>(xin, xout, Nin, np, ns,
                                                      rad * rad, idxb);
    long cols = (long)BATCH * np * ns;
    int Kp0 = (d4[0] + 31) & ~31;
    k_build_x_sa<<<(unsigned)((cols + 255) / 256), 256, 0, stream>>>(
        xin, xout, Fin, idxb, Nin, np, ns, Cin, Kp0, 1.f / rad, Xbuf);
    mlp_layer(Wp(wb, 0), d4[1], d4[0], cols);
    to_x(Gp(wb, 0), Bq(wb, 0), d4[1], cols);
    mlp_layer(Wp(wb, 1), d4[2], d4[1], cols);
    to_x(Gp(wb, 1), Bq(wb, 1), d4[2], cols);
    mlp_layer(Wp(wb, 2), d4[3], d4[2], cols);
    long tot = (long)BATCH * d4[3] * np;
    k_bn_relu_max<<<(unsigned)((tot + 255) / 256), 256, 0, stream>>>(
        Ybuf, meanb, varb, Gp(wb, 2), Bq(wb, 2), Fout, d4[3], np, ns);
  };

  auto run_fp = [&](const float* ux, int nu, const float* kx, int nk,
                    const float* UF, int Cu, const float* KF, int Ck,
                    int wb, float* Fout) {
    int tq = BATCH * nu;
    k_build_x_fp<<<(tq + 255) / 256, 256, 0, stream>>>(ux, kx, UF, KF, nu, nk,
                                                       Cu, Ck, Xbuf);
    long cols = (long)BATCH * nu;
    mlp_layer(Wp(wb, 0), 256, Ck + Cu, cols);
    to_x(Gp(wb, 0), Bq(wb, 0), 256, cols);
    mlp_layer(Wp(wb, 1), 256, 256, cols);
    long tot = 256 * cols;
    k_bn_relu_fp_out<<<(unsigned)((tot + 255) / 256), 256, 0, stream>>>(
        Ybuf, meanb, varb, Gp(wb, 1), Bq(wb, 1), Fout, 256, nu, cols);
  };

  // ---- pipeline ----
  k_split<<<(BATCH * N0 + 255) / 256, 256, 0, stream>>>(pc, xyz0, f0, N0);

  const int dsa1[4] = {4, 64, 64, 128};
  const int dsa2[4] = {131, 128, 128, 256};
  const int dsa3[4] = {259, 128, 128, 256};
  const int dsa4[4] = {259, 128, 128, 256};

  run_sa(xyz0, N0,   f0, 1,   2048, 64, 0.2f, sa1b, dsa1, xyz1, f1, inds1);
  run_sa(xyz1, 2048, f1, 128, 1024, 32, 0.4f, sa2b, dsa2, xyz2, f2, indsT);
  run_sa(xyz2, 1024, f2, 256, 512,  16, 0.8f, sa3b, dsa3, xyz3, f3, indsT);
  run_sa(xyz3, 512,  f3, 256, 256,  16, 1.2f, sa4b, dsa4, xyz4, f4, indsT);

  run_fp(xyz3, 512,  xyz4, 256, f3, 256, f4,   256, fp1b, fp1o);
  run_fp(xyz2, 1024, xyz3, 512, f2, 256, fp1o, 256, fp2b, (float*)d_out);

  k_tail<<<(16384 + 255) / 256, 256, 0, stream>>>(xyz2, inds1, (float*)d_out);
}
